// InnerPNN_41455024341615
// MI455X (gfx1250) — matrix operand order repaired and checked by reading, not compiled, not run on hardware
//
#include <hip/hip_runtime.h>
#include <math.h>

typedef __attribute__((ext_vector_type(16))) _Float16 v16h;
typedef __attribute__((ext_vector_type(8)))  _Float16 v8h;
typedef __attribute__((ext_vector_type(8)))  float    v8f;

#define BATCH 16384
#define NF 30
#define ED 64
#define DIN 2355
#define KP0 2368          // DIN padded to multiple of 32
#define APITCH 2376       // LDS row pitch (halves), keeps 16B alignment, de-strides banks

// ---------------- WMMA helpers ----------------

__device__ __forceinline__ v8f wmma16x16x32(v16h a, v16h b, v8f c) {
  // 8 args: (neg_a, A, neg_b, B, c_mod, C, reuse_a, reuse_b)
  return __builtin_amdgcn_wmma_f32_16x16x32_f16(false, a, false, b, (short)0, c,
                                                false, false);
}

// A fragment (16xK rows-of-samples), row-major source with pitch `ld` halves.
// ISA 16-bit A layout: lanes 0-15 hold K {0..7,16..23} of row m=lane,
// lanes 16-31 hold K {8..15,24..31} of row m=lane-16.
template <typename PtrT>
__device__ __forceinline__ v16h load_a_frag(const PtrT* base, int ld) {
  const int lane = threadIdx.x & 31;
  const int m  = lane & 15;
  const int ks = (lane >> 4) * 8;
  const PtrT* p = base + m * ld + ks;
  union { v16h v; v8h h[2]; } u;
  u.h[0] = *(const v8h*)(p);        // K = ks .. ks+7
  u.h[1] = *(const v8h*)(p + 16);   // K = ks+16 .. ks+23
  return u.v;
}

// B fragment from packed W^T (N x K, row-major, f16): lane n holds column n,
// lanes 0-15 -> K {0..7,16..23}, lanes 16-31 -> K {8..15,24..31} (matches A).
__device__ __forceinline__ v16h load_b_frag(const _Float16* __restrict__ wt,
                                            int K, int ntile, int kt) {
  const int lane = threadIdx.x & 31;
  const int n  = lane & 15;
  const int k0 = (lane >> 4) * 8;
  const _Float16* p = wt + (size_t)(ntile * 16 + n) * K + kt * 32 + k0;
  union { v16h v; v8h h[2]; } u;
  u.h[0] = *(const v8h*)(p);
  u.h[1] = *(const v8h*)(p + 16);
  return u.v;
}

// ---------------- weight transpose + f16 pack ----------------
// src: K x N (f32, row-major)  ->  dst: N x Kpad (f16), zero-padded K.
__global__ __launch_bounds__(256) void pack_wt(const float* __restrict__ src,
                                               _Float16* __restrict__ dst,
                                               int K, int Kpad, int N) {
  int idx = blockIdx.x * 256 + threadIdx.x;
  int total = N * Kpad;
  if (idx >= total) return;
  int nrow = idx / Kpad;
  int k = idx - nrow * Kpad;
  float v = (k < K) ? src[(size_t)k * N + nrow] : 0.0f;
  dst[idx] = (_Float16)v;
}

// ---------------- fused: gather + gram + (h @ W0 + b0).relu ----------------
// One block = 16 samples. LDS holds the 16 x 2368 f16 A-tile (never hits HBM).
__global__ __launch_bounds__(256) void fused_embed_gemm0(
    const int* __restrict__ x, const float* __restrict__ emb,
    const _Float16* __restrict__ wt0, const float* __restrict__ bias0,
    _Float16* __restrict__ h1) {
  __shared__ _Float16 As[16 * APITCH];  // 76,032 bytes
  const int tid = threadIdx.x;
  const int s0 = blockIdx.x * 16;

  // gather embeddings -> LDS (also the first F*D entries of h)
  for (int idx = tid; idx < 16 * NF * ED; idx += 256) {
    int s = idx / (NF * ED);
    int r = idx - s * (NF * ED);
    int f = r >> 6;
    int d = r & 63;
    int row = x[(s0 + s) * NF + f];
    As[s * APITCH + r] = (_Float16)emb[(size_t)row * ED + d];
  }
  __syncthreads();

  // cross terms: 435 upper-triangle dots per sample (+13 zero pads to 2368)
  for (int idx = tid; idx < 16 * 448; idx += 256) {
    int s = idx / 448;
    int p = idx - s * 448;
    if (p < 435) {
      int i = 0, rem = p;
      while (rem >= (NF - 1) - i) { rem -= (NF - 1) - i; ++i; }
      int j = i + 1 + rem;
      const _Float16* pi = &As[s * APITCH + i * ED];
      const _Float16* pj = &As[s * APITCH + j * ED];
      float acc = 0.0f;
      #pragma unroll 8
      for (int d = 0; d < ED; ++d) acc += (float)pi[d] * (float)pj[d];
      As[s * APITCH + NF * ED + p] = (_Float16)acc;
    } else {
      As[s * APITCH + NF * ED + p] = (_Float16)0.0f;  // K-pad 2355..2367
    }
  }
  __syncthreads();

  // GEMM0: 16 x 2368 (LDS) @ 2368 x 512 -> 16 x 512; 8 waves x 4 N-tiles
  const int lane = tid & 31;
  const int w = tid >> 5;
  v8f acc[4] = {};
  for (int kt = 0; kt < KP0 / 32; ++kt) {
    v16h a = load_a_frag(&As[kt * 32], APITCH);
    #pragma unroll
    for (int q = 0; q < 4; ++q) {
      v16h b = load_b_frag(wt0, KP0, w + q * 8, kt);
      acc[q] = wmma16x16x32(a, b, acc[q]);
    }
  }
  const int n = lane & 15;
  const int mhi = (lane >> 4) * 8;
  #pragma unroll
  for (int q = 0; q < 4; ++q) {
    int nt = w + q * 8;
    float bv = bias0[nt * 16 + n];
    #pragma unroll
    for (int v = 0; v < 8; ++v) {
      float val = acc[q][v] + bv;
      val = val > 0.0f ? val : 0.0f;
      h1[(size_t)(s0 + v + mhi) * 512 + nt * 16 + n] = (_Float16)val;
    }
  }
}

// ---------------- generic WMMA GEMM + bias + ReLU (f16 in/out) ----------------
template <int K, int N>
__global__ __launch_bounds__(256) void gemm_relu(
    const _Float16* __restrict__ A, const _Float16* __restrict__ WT,
    const float* __restrict__ bias, _Float16* __restrict__ out) {
  const int lane = threadIdx.x & 31;
  const int w = threadIdx.x >> 5;
  const int mtile = blockIdx.x * 8 + w;
  const int ntile = blockIdx.y;
  const _Float16* abase = A + (size_t)mtile * 16 * K;
  v8f acc = {};
  #pragma unroll 4
  for (int kt = 0; kt < K / 32; ++kt) {
    v16h a = load_a_frag(abase + kt * 32, K);
    v16h b = load_b_frag(WT, K, ntile, kt);
    acc = wmma16x16x32(a, b, acc);
  }
  const int n = lane & 15;
  const int mhi = (lane >> 4) * 8;
  float bv = bias[ntile * 16 + n];
  #pragma unroll
  for (int v = 0; v < 8; ++v) {
    float val = acc[v] + bv;
    val = val > 0.0f ? val : 0.0f;
    out[(size_t)(mtile * 16 + v + mhi) * N + ntile * 16 + n] = (_Float16)val;
  }
}

// ---------------- final 64 -> 1 + sigmoid (f32) ----------------
__global__ __launch_bounds__(256) void final_sigmoid(
    const _Float16* __restrict__ h4, const float* __restrict__ W4,
    const float* __restrict__ b4, float* __restrict__ out) {
  int b = blockIdx.x * 256 + threadIdx.x;
  if (b >= BATCH) return;
  const _Float16* row = h4 + (size_t)b * 64;
  float s = b4[0];
  #pragma unroll
  for (int d = 0; d < 64; ++d) s += (float)row[d] * W4[d];
  out[b] = 1.0f / (1.0f + expf(-s));
}

// ---------------- launcher ----------------
extern "C" void kernel_launch(void* const* d_in, const int* in_sizes, int n_in,
                              void* d_out, int out_size, void* d_ws, size_t ws_size,
                              hipStream_t stream) {
  const int*   x   = (const int*)  d_in[0];
  const float* emb = (const float*)d_in[1];
  const float* W0  = (const float*)d_in[2];  const float* b0 = (const float*)d_in[3];
  const float* W1  = (const float*)d_in[4];  const float* b1 = (const float*)d_in[5];
  const float* W2  = (const float*)d_in[6];  const float* b2 = (const float*)d_in[7];
  const float* W3  = (const float*)d_in[8];  const float* b3 = (const float*)d_in[9];
  const float* W4  = (const float*)d_in[10]; const float* b4 = (const float*)d_in[11];
  float* out = (float*)d_out;

  char* ws = (char*)d_ws;
  size_t off = 0;
  auto carve = [&](size_t bytes) {
    void* p = ws + off;
    off = (off + bytes + 255) & ~(size_t)255;
    return p;
  };
  _Float16* wt0 = (_Float16*)carve((size_t)512 * KP0 * 2);
  _Float16* wt1 = (_Float16*)carve((size_t)256 * 512 * 2);
  _Float16* wt2 = (_Float16*)carve((size_t)128 * 256 * 2);
  _Float16* wt3 = (_Float16*)carve((size_t)64  * 128 * 2);
  _Float16* h1  = (_Float16*)carve((size_t)BATCH * 512 * 2);
  _Float16* h2  = (_Float16*)carve((size_t)BATCH * 256 * 2);
  _Float16* h3  = (_Float16*)carve((size_t)BATCH * 128 * 2);
  _Float16* h4  = (_Float16*)carve((size_t)BATCH * 64  * 2);

  // pack weights (transpose to N x Kpad, f16)
  pack_wt<<<(512 * KP0 + 255) / 256, 256, 0, stream>>>(W0, wt0, DIN, KP0, 512);
  pack_wt<<<(256 * 512 + 255) / 256, 256, 0, stream>>>(W1, wt1, 512, 512, 256);
  pack_wt<<<(128 * 256 + 255) / 256, 256, 0, stream>>>(W2, wt2, 256, 256, 128);
  pack_wt<<<(64  * 128 + 255) / 256, 256, 0, stream>>>(W3, wt3, 128, 128, 64);

  // fused gather + gram + layer0
  fused_embed_gemm0<<<BATCH / 16, 256, 0, stream>>>(x, emb, wt0, b0, h1);

  // MLP tail
  gemm_relu<512, 256><<<dim3(BATCH / 128, 256 / 16), 256, 0, stream>>>(h1, wt1, b1, h2);
  gemm_relu<256, 128><<<dim3(BATCH / 128, 128 / 16), 256, 0, stream>>>(h2, wt2, b2, h3);
  gemm_relu<128, 64> <<<dim3(BATCH / 128, 64 / 16),  256, 0, stream>>>(h3, wt3, b3, h4);

  final_sigmoid<<<BATCH / 256, 256, 0, stream>>>(h4, W4, b4, out);
}
